// GIN_87024627352319
// MI455X (gfx1250) — matrix operand order repaired and checked by reading, not compiled, not run on hardware
//
#include <hip/hip_runtime.h>
#include <hip/hip_bf16.h>

// Problem constants (match reference)
#define NN   50000
#define EE   800000
#define CH   128      // in/out channels
#define LL   3
#define CATW 512      // 128 + 3*128
#define BN_EPS 1e-5f

typedef __attribute__((ext_vector_type(16))) __bf16 v16bf;
typedef __attribute__((ext_vector_type(8)))  float  v8f;

static __device__ __forceinline__ unsigned short f2bf(float f) {
    unsigned int u = __float_as_uint(f);
    u += 0x7FFFu + ((u >> 16) & 1u);          // round-to-nearest-even
    return (unsigned short)(u >> 16);
}

// ---------------------------------------------------------------------------
// Weight prep: fp32 [K,N] -> bf16 [N,K] (transposed so B-fragments are contiguous)
// W1,W2: [L][128][128]; finW: [512][128]
// ---------------------------------------------------------------------------
__global__ void prep_weights(const float* __restrict__ W1, const float* __restrict__ W2,
                             const float* __restrict__ finW,
                             unsigned short* __restrict__ W1t, unsigned short* __restrict__ W2t,
                             unsigned short* __restrict__ finWt) {
    int idx = blockIdx.x * blockDim.x + threadIdx.x;
    if (idx < LL * CH * CH) {
        int l = idx / (CH * CH);
        int r = idx % (CH * CH);
        int k = r / CH, n = r % CH;
        W1t[l * CH * CH + n * CH + k] = f2bf(W1[idx]);
        W2t[l * CH * CH + n * CH + k] = f2bf(W2[idx]);
    }
    if (idx < CATW * CH) {
        int k = idx / CH, n = idx % CH;
        finWt[n * CATW + k] = f2bf(finW[idx]);
    }
}

// h = x (fp32 copy) and xcat[:, 0:128] = bf16(x)
__global__ void init_h_xcat(const float* __restrict__ x, float* __restrict__ h,
                            unsigned short* __restrict__ xcat) {
    size_t total = (size_t)NN * CH;
    for (size_t i = (size_t)blockIdx.x * blockDim.x + threadIdx.x; i < total;
         i += (size_t)gridDim.x * blockDim.x) {
        float v = x[i];
        h[i] = v;
        size_t row = i >> 7; int c = (int)(i & 127);
        xcat[row * CATW + c] = f2bf(v);
    }
}

// agg = (1 + eps[l]) * h
__global__ void agg_init(const float* __restrict__ h, float* __restrict__ agg,
                         const float* __restrict__ eps, int l) {
    float s = 1.0f + eps[l];
    size_t total = (size_t)NN * CH;
    for (size_t i = (size_t)blockIdx.x * blockDim.x + threadIdx.x; i < total;
         i += (size_t)gridDim.x * blockDim.x)
        agg[i] = s * h[i];
}

// scatter-add: agg[dst[e], c] += h[src[e], c]   (L2-resident fp32 atomics)
__global__ void scatter_add(const float* __restrict__ h, const int* __restrict__ src,
                            const int* __restrict__ dst, float* __restrict__ agg) {
    size_t total = (size_t)EE * CH;
    for (size_t t = (size_t)blockIdx.x * blockDim.x + threadIdx.x; t < total;
         t += (size_t)gridDim.x * blockDim.x) {
        size_t e = t >> 7; int c = (int)(t & 127);
        int s = src[e], d = dst[e];
        atomicAdd(&agg[(size_t)d * CH + c], h[(size_t)s * CH + c]);
    }
}

// fp32 -> bf16 elementwise
__global__ void to_bf16(const float* __restrict__ in, unsigned short* __restrict__ out, size_t total) {
    for (size_t i = (size_t)blockIdx.x * blockDim.x + threadIdx.x; i < total;
         i += (size_t)gridDim.x * blockDim.x)
        out[i] = f2bf(in[i]);
}

// ---------------------------------------------------------------------------
// WMMA GEMM: C[M,128] = relu?( A[M,K](bf16) @ Bt[128,K]^T(bf16) + bias )
// One 16x16 tile per wave; 8 waves/block cover the 128-wide output strip.
// M is a multiple of 16, K a multiple of 32 (128 or 512).
// ---------------------------------------------------------------------------
__global__ __launch_bounds__(256)
void gemm_bf16_wmma(const unsigned short* __restrict__ A,
                    const unsigned short* __restrict__ Bt,
                    const float* __restrict__ bias,
                    float* __restrict__ C, int K, int relu) {
    int wave = threadIdx.x >> 5;       // N-tile 0..7
    int lane = threadIdx.x & 31;
    int grp  = lane >> 4;              // 0/1: lane group selects K window
    int lr   = lane & 15;
    int tile_m = blockIdx.x;           // 0..M/16-1
    int m = tile_m * 16 + lr;          // A row this lane feeds
    int n = wave * 16 + lr;            // B column this lane feeds == C column

    const unsigned short* arow = A  + (size_t)m * K;
    const unsigned short* brow = Bt + (size_t)n * K;

    v8f acc = {};
    for (int kb = 0; kb < K; kb += 32) {
        union { uint4 u[2]; v16bf v; } a, b;
        // A 16-bit layout: elems 0..7 -> K = kb+grp*8+e ; elems 8..15 -> K = kb+16+grp*8+(e-8)
        a.u[0] = *(const uint4*)(arow + kb + grp * 8);
        a.u[1] = *(const uint4*)(arow + kb + 16 + grp * 8);
        // B layout: same as A: elems 0..7 -> K = kb+grp*8+e ; elems 8..15 -> K = kb+16+grp*8+(e-8)
        b.u[0] = *(const uint4*)(brow + kb + grp * 8);
        b.u[1] = *(const uint4*)(brow + kb + 16 + grp * 8);
        acc = __builtin_amdgcn_wmma_f32_16x16x32_bf16(
            false, a.v, false, b.v, (short)0, acc, false, false);
    }

    float bv = bias[n];
#pragma unroll
    for (int r = 0; r < 8; ++r) {      // C/D: lane L, vgpr r -> (M=r+8*grp, N=lr)
        int mm = tile_m * 16 + r + grp * 8;
        float v = acc[r] + bv;
        if (relu) v = fmaxf(v, 0.0f);
        C[(size_t)mm * CH + n] = v;
    }
}

// ---------------------------------------------------------------------------
// BatchNorm (training-mode batch statistics)
// ---------------------------------------------------------------------------
__global__ void zero_stats(float* __restrict__ stats) {
    if (threadIdx.x < 256) stats[threadIdx.x] = 0.0f;
}

// stats[0:128]=sum, stats[128:256]=sumsq ; thread owns one column per row-partition
__global__ void col_stats(const float* __restrict__ Y, float* __restrict__ stats) {
    int c = threadIdx.x & 127;
    int part   = (int)((blockIdx.x * blockDim.x + threadIdx.x) >> 7);
    int nparts = (int)((gridDim.x * blockDim.x) >> 7);
    float s = 0.0f, q = 0.0f;
    for (int r = part; r < NN; r += nparts) {
        float v = Y[(size_t)r * CH + c];
        s += v; q += v * v;
    }
    atomicAdd(&stats[c], s);
    atomicAdd(&stats[128 + c], q);
}

// scale/shift folding gamma,beta,mu,var
__global__ void bn_finalize(const float* __restrict__ stats, const float* __restrict__ gamma,
                            const float* __restrict__ beta, float* __restrict__ ss) {
    int c = threadIdx.x;
    if (c < 128) {
        float inv = 1.0f / (float)NN;
        float mu  = stats[c] * inv;
        float var = stats[128 + c] * inv - mu * mu;
        float sc  = gamma[c] * rsqrtf(var + BN_EPS);
        ss[c] = sc;
        ss[128 + c] = beta[c] - mu * sc;
    }
}

// normalized -> bf16 only (feeds next GEMM)
__global__ void bn_apply_bf16(const float* __restrict__ Y, const float* __restrict__ ss,
                              unsigned short* __restrict__ outb) {
    size_t total = (size_t)NN * CH;
    for (size_t i = (size_t)blockIdx.x * blockDim.x + threadIdx.x; i < total;
         i += (size_t)gridDim.x * blockDim.x) {
        int c = (int)(i & 127);
        outb[i] = f2bf(Y[i] * ss[c] + ss[128 + c]);
    }
}

// normalized -> fp32 h (next layer input) + bf16 slice of concat buffer
__global__ void bn_apply_dual(const float* __restrict__ Y, const float* __restrict__ ss,
                              float* __restrict__ h, unsigned short* __restrict__ xcat,
                              int coloff) {
    size_t total = (size_t)NN * CH;
    for (size_t i = (size_t)blockIdx.x * blockDim.x + threadIdx.x; i < total;
         i += (size_t)gridDim.x * blockDim.x) {
        int c = (int)(i & 127);
        float v = Y[i] * ss[c] + ss[128 + c];
        h[i] = v;
        size_t row = i >> 7;
        xcat[row * CATW + coloff + c] = f2bf(v);
    }
}

// ---------------------------------------------------------------------------
extern "C" void kernel_launch(void* const* d_in, const int* in_sizes, int n_in,
                              void* d_out, int out_size, void* d_ws, size_t ws_size,
                              hipStream_t stream) {
    (void)in_sizes; (void)n_in; (void)out_size; (void)ws_size;

    const float* x    = (const float*)d_in[0];
    const int*   ei   = (const int*)d_in[1];   // [2][E]
    const float* W1   = (const float*)d_in[2];
    const float* b1   = (const float*)d_in[3];
    const float* g1   = (const float*)d_in[4];
    const float* be1  = (const float*)d_in[5];
    const float* W2   = (const float*)d_in[6];
    const float* b2   = (const float*)d_in[7];
    const float* g2   = (const float*)d_in[8];
    const float* be2  = (const float*)d_in[9];
    const float* eps  = (const float*)d_in[10];
    const float* finW = (const float*)d_in[11];
    const float* finb = (const float*)d_in[12];
    float* out = (float*)d_out;

    const int* src = ei;
    const int* dst = ei + EE;

    // workspace carving (256B aligned)
    char* p = (char*)d_ws;
    auto carve = [&p](size_t bytes) -> void* {
        void* r = (void*)p;
        p += (bytes + 255) & ~(size_t)255;
        return r;
    };
    float*          h     = (float*)carve((size_t)NN * CH * 4);          // layer input fp32
    float*          Y     = (float*)carve((size_t)NN * CH * 4);          // agg / GEMM output
    unsigned short* zb    = (unsigned short*)carve((size_t)NN * CH * 2); // bf16 GEMM A
    unsigned short* xcat  = (unsigned short*)carve((size_t)NN * CATW * 2);
    unsigned short* W1t   = (unsigned short*)carve((size_t)LL * CH * CH * 2);
    unsigned short* W2t   = (unsigned short*)carve((size_t)LL * CH * CH * 2);
    unsigned short* finWt = (unsigned short*)carve((size_t)CATW * CH * 2);
    float*          stats = (float*)carve(256 * 4);
    float*          ss    = (float*)carve(256 * 4);

    const int EW = 4096, TB = 256;          // elementwise grid
    const int MT = NN / 16;                 // 3125 row tiles

    prep_weights<<<(CATW * CH + TB - 1) / TB, TB, 0, stream>>>(W1, W2, finW, W1t, W2t, finWt);
    init_h_xcat<<<EW, TB, 0, stream>>>(x, h, xcat);

    for (int l = 0; l < LL; ++l) {
        // GIN aggregation: z = (1+eps)*h + segment_sum(h[src] -> dst)
        agg_init<<<EW, TB, 0, stream>>>(h, Y, eps, l);
        scatter_add<<<EW, TB, 0, stream>>>(h, src, dst, Y);
        to_bf16<<<EW, TB, 0, stream>>>(Y, zb, (size_t)NN * CH);

        // Lin1 + ReLU  (writes over Y, reads zb)
        gemm_bf16_wmma<<<MT, 256, 0, stream>>>(zb, W1t + (size_t)l * CH * CH,
                                               b1 + l * CH, Y, CH, 1);
        // BN1
        zero_stats<<<1, 256, 0, stream>>>(stats);
        col_stats<<<512, 256, 0, stream>>>(Y, stats);
        bn_finalize<<<1, 128, 0, stream>>>(stats, g1 + l * CH, be1 + l * CH, ss);
        bn_apply_bf16<<<EW, TB, 0, stream>>>(Y, ss, zb);

        // Lin2 + ReLU
        gemm_bf16_wmma<<<MT, 256, 0, stream>>>(zb, W2t + (size_t)l * CH * CH,
                                               b2 + l * CH, Y, CH, 1);
        // BN2 -> new h (fp32) + concat slice (bf16)
        zero_stats<<<1, 256, 0, stream>>>(stats);
        col_stats<<<512, 256, 0, stream>>>(Y, stats);
        bn_finalize<<<1, 128, 0, stream>>>(stats, g2 + l * CH, be2 + l * CH, ss);
        bn_apply_dual<<<EW, TB, 0, stream>>>(Y, ss, h, xcat, CH * (l + 1));
    }

    // out = concat(x, z1, z2, z3) @ finW + finb   (K = 512, no relu)
    gemm_bf16_wmma<<<MT, 256, 0, stream>>>(xcat, finWt, finb, out, CATW, 0);
}
